// PersonalizedProbabilityCalculator_89507118449342
// MI455X (gfx1250) — matrix operand order repaired and checked by reading, not compiled, not run on hardware
//
#include <hip/hip_runtime.h>
#include <hip/hip_bf16.h>

#define NUM_ITEMS 10000
#define EMBED_DIM 64
#define BATCH 32
#define MAX_HIST 200
#define NPAD 224               // padded history length (7 * 32)
#define NT 14                  // 16-wide n-tiles in NPAD
#define LEAKY 0.2f
#define WUSER 0.5f

#define H_STRIDE 72            // bf16 elems per sH row (144B, conflict-free b128 loads)
#define HT_STRIDE 240          // bf16 elems per sHt row (480B, conflict-free b128 loads)

typedef __attribute__((ext_vector_type(16))) __bf16 v16bf;
typedef __attribute__((ext_vector_type(8)))  __bf16 v8bf;
typedef __attribute__((ext_vector_type(8)))  float  v8f;

// cross-half (xor 16) lane exchange within wave32 via ds_swizzle group-of-32
#define SWZ(x, imm) __int_as_float(__builtin_amdgcn_ds_swizzle(__float_as_int(x), (imm)))
#define XOR16(x) SWZ((x), 0x401F)

// ---------------------------------------------------------------------------
// Precompute: G = E @ W  ([I,F]), c = E @ bias ([I])  -> folds away the final
// projection GEMM: out[b,i] = dot(P[i,:], G[i,:]) + c[i]
// ---------------------------------------------------------------------------
__global__ void prep_G_kernel(const float* __restrict__ E,
                              const float* __restrict__ W,
                              const float* __restrict__ bias,
                              float* __restrict__ G,
                              float* __restrict__ cvec) {
    __shared__ float sE[EMBED_DIM];
    __shared__ float sP[EMBED_DIM];
    const int i = blockIdx.x;
    const int t = threadIdx.x;
    float e = E[(size_t)i * EMBED_DIM + t];
    sE[t] = e;
    sP[t] = e * bias[t];
    __syncthreads();
    float acc = 0.f;
#pragma unroll
    for (int f = 0; f < EMBED_DIM; ++f) acc += sE[f] * W[f * EMBED_DIM + t];
    G[(size_t)i * EMBED_DIM + t] = acc;
    if (t == 0) {
        float cs = 0.f;
#pragma unroll
        for (int f = 0; f < EMBED_DIM; ++f) cs += sP[f];
        cvec[i] = cs;
    }
}

// ---------------------------------------------------------------------------
// Fused attention kernel: one workgroup = (batch b, 64-item chunk), 4 waves.
// Per wave (16 items):  S^T = H @ E^T (WMMA bf16) -> leaky/mask/softmax in
// registers -> blend with user scores -> P^T = H^T @ attn^T (WMMA bf16) ->
// out = rowdot(P, G) + c.
// ---------------------------------------------------------------------------
__global__ __launch_bounds__(128) void
persattn_kernel(const int* __restrict__ ids,
                const int* __restrict__ lens,
                const float* __restrict__ E,
                const float* __restrict__ U,
                const float* __restrict__ G,
                const float* __restrict__ cvec,
                float* __restrict__ out) {
    __shared__ __align__(16) __bf16 sH[NPAD * H_STRIDE];       // history rows [n][f]
    __shared__ __align__(16) __bf16 sHt[EMBED_DIM * HT_STRIDE]; // transposed [f][n]
    __shared__ float sUser[NPAD];                               // 0.5 * user softmax

    const int b   = blockIdx.y;
    const int tid = threadIdx.x;
    const int len = lens[b];

    // ---- gather history, convert fp32 -> bf16, fill sH and sHt ----
    for (int n = tid; n < NPAD; n += 128) {
        if (n < MAX_HIST) {
            const int id = ids[b * MAX_HIST + n];
            const float4* src = (const float4*)(E + (size_t)id * EMBED_DIM);
#pragma unroll
            for (int q = 0; q < 16; ++q) {
                float4 v = src[q];
                __bf16 b0 = (__bf16)v.x, b1 = (__bf16)v.y, b2 = (__bf16)v.z, b3 = (__bf16)v.w;
                sH[n * H_STRIDE + 4 * q + 0] = b0;
                sH[n * H_STRIDE + 4 * q + 1] = b1;
                sH[n * H_STRIDE + 4 * q + 2] = b2;
                sH[n * H_STRIDE + 4 * q + 3] = b3;
                sHt[(4 * q + 0) * HT_STRIDE + n] = b0;
                sHt[(4 * q + 1) * HT_STRIDE + n] = b1;
                sHt[(4 * q + 2) * HT_STRIDE + n] = b2;
                sHt[(4 * q + 3) * HT_STRIDE + n] = b3;
            }
        } else {
            __bf16 z = (__bf16)0.f;
#pragma unroll
            for (int f = 0; f < EMBED_DIM; ++f) {
                sH[n * H_STRIDE + f] = z;
                sHt[f * HT_STRIDE + n] = z;
            }
        }
    }
    __syncthreads();

    // ---- user-perspective logits (leaky + mask) ----
    for (int n = tid; n < NPAD; n += 128) {
        float d;
        if (n < len) {
            d = 0.f;
#pragma unroll
            for (int f = 0; f < EMBED_DIM; ++f)
                d += U[f] * (float)sH[n * H_STRIDE + f];
            d = (d > 0.f) ? d : LEAKY * d;
        } else {
            d = -__builtin_inff();
        }
        sUser[n] = d;
    }
    __syncthreads();

    // ---- user softmax (wave 0), store 0.5 * score ----
    if (tid < 32) {
        float m = -__builtin_inff();
        for (int n = tid; n < NPAD; n += 32) m = fmaxf(m, sUser[n]);
        m = fmaxf(m, SWZ(m, 0x041F));
        m = fmaxf(m, SWZ(m, 0x081F));
        m = fmaxf(m, SWZ(m, 0x101F));
        m = fmaxf(m, SWZ(m, 0x201F));
        m = fmaxf(m, SWZ(m, 0x401F));
        float s = 0.f;
        for (int n = tid; n < NPAD; n += 32) s += __expf(sUser[n] - m);
        s += SWZ(s, 0x041F);
        s += SWZ(s, 0x081F);
        s += SWZ(s, 0x101F);
        s += SWZ(s, 0x201F);
        s += SWZ(s, 0x401F);
        const float rinv = WUSER / s;
        for (int n = tid; n < NPAD; n += 32)
            sUser[n] = __expf(sUser[n] - m) * rinv;   // exp(-inf)=0 for invalid n
    }
    __syncthreads();

    // ---- per-wave fused attention over a 16-item tile ----
    const int wave = tid >> 5;
    const int lane = tid & 31;
    const int tile = blockIdx.x * 4 + wave;
    if (tile * 16 >= NUM_ITEMS) return;           // uniform per wave, after all barriers

    const int  li    = lane & 15;
    const bool hihalf = (lane >= 16);
    const int  item  = tile * 16 + li;
    const int  anoff = hihalf ? 16 : 0;           // A-operand K-chunk offset for this half

    // B-operand of GEMM1: E^T slices (K = f).  B[kk, col=item]; per lane 16
    // consecutive f values of its item's row, for kb = 0 and 32.
    v16bf eB[2];
    {
        const float* erow = E + (size_t)item * EMBED_DIM + (hihalf ? 16 : 0);
#pragma unroll
        for (int kb = 0; kb < 2; ++kb) {
            const float4* p = (const float4*)(erow + 32 * kb);
#pragma unroll
            for (int q = 0; q < 4; ++q) {
                float4 v = p[q];
                eB[kb][4 * q + 0] = (__bf16)v.x;
                eB[kb][4 * q + 1] = (__bf16)v.y;
                eB[kb][4 * q + 2] = (__bf16)v.z;
                eB[kb][4 * q + 3] = (__bf16)v.w;
            }
        }
    }

    // GEMM1: S^T tiles.  s[t][v] = S^T(n = 16t + v + 8*hihalf, item = li)
    float s[NT][8];
#pragma unroll
    for (int t = 0; t < NT; ++t) {
        v8f acc = {};
#pragma unroll
        for (int kb = 0; kb < 2; ++kb) {
            const __bf16* hp = &sH[(16 * t + li) * H_STRIDE + 32 * kb + anoff];
            v8bf lo = *(const v8bf*)hp;
            v8bf hi = *(const v8bf*)(hp + 8);
            v16bf A = __builtin_shufflevector(lo, hi, 0, 1, 2, 3, 4, 5, 6, 7,
                                                      8, 9, 10, 11, 12, 13, 14, 15);
            acc = __builtin_amdgcn_wmma_f32_16x16x32_bf16(
                false, A, false, eB[kb], (short)0, acc, false, false);
        }
#pragma unroll
        for (int v = 0; v < 8; ++v) s[t][v] = acc[v];
    }

    // ---- masked leaky-ReLU + in-register softmax over n (one xor-16 exchange) ----
    float vmax = -__builtin_inff();
#pragma unroll
    for (int t = 0; t < NT; ++t)
#pragma unroll
        for (int v = 0; v < 8; ++v) {
            const int n = 16 * t + v + (hihalf ? 8 : 0);
            float x = s[t][v];
            x = (x > 0.f) ? x : LEAKY * x;
            x = (n < len) ? x : -__builtin_inff();
            s[t][v] = x;
            vmax = fmaxf(vmax, x);
        }
    vmax = fmaxf(vmax, XOR16(vmax));

    float vsum = 0.f;
#pragma unroll
    for (int t = 0; t < NT; ++t)
#pragma unroll
        for (int v = 0; v < 8; ++v) {
            float e = __expf(s[t][v] - vmax);
            s[t][v] = e;
            vsum += e;
        }
    vsum += XOR16(vsum);
    const float rinv = (1.0f - WUSER) / vsum;

#pragma unroll
    for (int t = 0; t < NT; ++t)
#pragma unroll
        for (int v = 0; v < 8; ++v) {
            const int n = 16 * t + v + (hihalf ? 8 : 0);
            s[t][v] = s[t][v] * rinv + sUser[n];   // blended attention weight
        }

    // GEMM2: P^T = H^T @ attn^T, K = n (7 steps of 32), 4 f-tiles.
    v8f pacc[4];
#pragma unroll
    for (int ft = 0; ft < 4; ++ft) pacc[ft] = (v8f){};

#pragma unroll
    for (int st = 0; st < 7; ++st) {
        // Assemble B operand (attn^T 32x16) from registers: one xor-16 exchange
        // swaps the half-tile each lane is missing.
        v16bf Bv;
#pragma unroll
        for (int j = 0; j < 8; ++j) {
            const float own0 = s[2 * st][j];       // n = 32*st + j + 8*hihalf
            const float own1 = s[2 * st + 1][j];   // n = 32*st + 16 + j + 8*hihalf
            const float sel  = hihalf ? own0 : own1;
            const float got  = XOR16(sel);
            const float e0 = hihalf ? got  : own0; // elem j    : n = base + j
            const float e1 = hihalf ? own1 : got;  // elem j+8  : n = base + 8 + j
            Bv[j]     = (__bf16)e0;
            Bv[j + 8] = (__bf16)e1;
        }
#pragma unroll
        for (int ft = 0; ft < 4; ++ft) {
            const __bf16* hp = &sHt[(16 * ft + li) * HT_STRIDE + 32 * st + anoff];
            v8bf lo = *(const v8bf*)hp;
            v8bf hi = *(const v8bf*)(hp + 8);
            v16bf A = __builtin_shufflevector(lo, hi, 0, 1, 2, 3, 4, 5, 6, 7,
                                                      8, 9, 10, 11, 12, 13, 14, 15);
            pacc[ft] = __builtin_amdgcn_wmma_f32_16x16x32_bf16(
                false, A, false, Bv, (short)0, pacc[ft], false, false);
        }
    }

    // ---- epilogue: out[b,i] = dot(P[i,:], G[i,:]) + c[i] ----
    const float* grow = G + (size_t)item * EMBED_DIM;
    float dot = 0.f;
#pragma unroll
    for (int ft = 0; ft < 4; ++ft)
#pragma unroll
        for (int v = 0; v < 8; ++v)
            dot += pacc[ft][v] * grow[16 * ft + v + (hihalf ? 8 : 0)];
    dot += XOR16(dot);

    if (!hihalf)
        out[(size_t)b * NUM_ITEMS + item] = dot + cvec[item];
}

// ---------------------------------------------------------------------------
extern "C" void kernel_launch(void* const* d_in, const int* in_sizes, int n_in,
                              void* d_out, int out_size, void* d_ws, size_t ws_size,
                              hipStream_t stream) {
    const int*   ids  = (const int*)d_in[0];    // [B, MAX_HIST]
    const int*   lens = (const int*)d_in[1];    // [B]
    const float* E    = (const float*)d_in[2];  // [I, F]
    const float* U    = (const float*)d_in[3];  // [1, F]
    const float* W    = (const float*)d_in[4];  // [F, F]
    const float* bias = (const float*)d_in[5];  // [F]
    float* out = (float*)d_out;                 // [B, I]

    float* G    = (float*)d_ws;                 // [I, F]
    float* cvec = G + (size_t)NUM_ITEMS * EMBED_DIM;  // [I]

    prep_G_kernel<<<NUM_ITEMS, EMBED_DIM, 0, stream>>>(E, W, bias, G, cvec);

    const int tiles = NUM_ITEMS / 16;           // 625
    dim3 grid((tiles + 3) / 4, BATCH);
    persattn_kernel<<<grid, 128, 0, stream>>>(ids, lens, E, U, G, cvec, out);
}
